// ChannelAttentionModule_62423054680110
// MI455X (gfx1250) — matrix operand order repaired and checked by reading, not compiled, not run on hardware
//
#include <hip/hip_runtime.h>
#include <hip/hip_bf16.h>
#include <cstddef>

typedef __attribute__((ext_vector_type(16))) __bf16 v16bf;
typedef __attribute__((ext_vector_type(8)))  float  v8f;

#define C_DIM 128
#define HW    16384
#define BATCH 32
#define CHUNK 64
#define LDSK  66          // CHUNK + 2 pad -> row stride 33 dwords, conflict-free
#define SEG   16
#define KSEG  (HW / SEG)  // 1024 K per workgroup
#define NT    64
#define LDSA  130         // 128 + 2 pad -> row stride 65 dwords, conflict-free

union Frag { unsigned u[8]; v16bf v; };

static __device__ __forceinline__ unsigned f2bf(float f) {
  unsigned u = __builtin_bit_cast(unsigned, f);
  u += 0x7FFFu + ((u >> 16) & 1u);   // round-to-nearest-even to bf16
  return u >> 16;
}

// ---------------------------------------------------------------------------
// Kernel 1a: partial Gram matrix S += F_seg . F_seg^T  (bf16 WMMA, f32 acc)
// grid = (SEG, BATCH), block = 256 (8 waves).
// Wave tiling: 2x4 tiles per wave (waves in a 4x2 grid over the 8x8 tiles).
// Double-buffered LDS staging: global float4 loads for chunk i+1 issue before
// the WMMA burst on chunk i; one barrier per chunk.
// ---------------------------------------------------------------------------
__global__ __launch_bounds__(256) void gram_partial_kernel(
    const float* __restrict__ x, float* __restrict__ scores) {
  __shared__ unsigned short lds[2][C_DIM * LDSK];
  const int seg  = blockIdx.x;
  const int b    = blockIdx.y;
  const int tid  = threadIdx.x;
  const int lane = tid & 31;
  const int wave = tid >> 5;
  const int l15  = lane & 15;
  const int lhi  = lane >> 4;
  const int bm   = wave >> 1;          // 0..3 -> tm in {2bm, 2bm+1}
  const int bn   = wave & 1;           // 0..1 -> tn in {4bn .. 4bn+3}
  const float* xb = x + (size_t)b * C_DIM * HW;

  v8f acc[2][4];
  #pragma unroll
  for (int i = 0; i < 2; ++i)
    #pragma unroll
    for (int j = 0; j < 4; ++j)
      #pragma unroll
      for (int e = 0; e < 8; ++e) acc[i][j][e] = 0.0f;

  // register staging: 8 float4 per thread = one 128x64 chunk for the block
  float4 st[8];
  const int srow = tid >> 4;           // 0..15 (+16 per pass)
  const int scol = (tid & 15) * 4;

  const int kbeg = seg * KSEG;
  const int nchunks = KSEG / CHUNK;

  auto load_chunk = [&](int k0) {
    #pragma unroll
    for (int p = 0; p < 8; ++p) {
      const int c = p * 16 + srow;
      st[p] = *reinterpret_cast<const float4*>(xb + (size_t)c * HW + k0 + scol);
    }
  };
  auto store_chunk = [&](int buf) {
    #pragma unroll
    for (int p = 0; p < 8; ++p) {
      const int c = p * 16 + srow;
      const unsigned pk0 = f2bf(st[p].x) | (f2bf(st[p].y) << 16);
      const unsigned pk1 = f2bf(st[p].z) | (f2bf(st[p].w) << 16);
      *reinterpret_cast<unsigned*>(&lds[buf][c * LDSK + scol])     = pk0;
      *reinterpret_cast<unsigned*>(&lds[buf][c * LDSK + scol + 2]) = pk1;
    }
  };
  auto compute_chunk = [&](int buf) {
    #pragma unroll
    for (int ks = 0; ks < CHUNK; ks += 32) {
      Frag A[2];
      #pragma unroll
      for (int i = 0; i < 2; ++i)
        #pragma unroll
        for (int v = 0; v < 8; ++v) {
          const int kk = ks + lhi * 8 + (v & 3) * 2 + (v >> 2) * 16;
          A[i].u[v] = *reinterpret_cast<const unsigned*>(
              &lds[buf][(16 * (2 * bm + i) + l15) * LDSK + kk]);
        }
      #pragma unroll
      for (int j = 0; j < 4; ++j) {
        Frag Bf;
        #pragma unroll
        for (int v = 0; v < 8; ++v) {
          const int kk = ks + lhi * 8 + (v & 3) * 2 + (v >> 2) * 16;
          Bf.u[v] = *reinterpret_cast<const unsigned*>(
              &lds[buf][(16 * (4 * bn + j) + l15) * LDSK + kk]);
        }
        #pragma unroll
        for (int i = 0; i < 2; ++i)
          acc[i][j] = __builtin_amdgcn_wmma_f32_16x16x32_bf16(
              false, A[i].v, false, Bf.v, (short)0, acc[i][j], false, false);
      }
    }
  };

  // prologue
  load_chunk(kbeg);
  store_chunk(0);
  __syncthreads();
  for (int ci = 0; ci < nchunks; ++ci) {
    const int cur = ci & 1;
    if (ci + 1 < nchunks) load_chunk(kbeg + (ci + 1) * CHUNK);  // prefetch
    compute_chunk(cur);
    if (ci + 1 < nchunks) store_chunk(cur ^ 1);
    __syncthreads();
  }

  float* sb = scores + (size_t)b * C_DIM * C_DIM;
  #pragma unroll
  for (int i = 0; i < 2; ++i)
    #pragma unroll
    for (int j = 0; j < 4; ++j)
      #pragma unroll
      for (int r = 0; r < 8; ++r) {
        const int row = 16 * (2 * bm + i) + r + 8 * lhi;
        const int col = 16 * (4 * bn + j) + l15;
        unsafeAtomicAdd(&sb[row * C_DIM + col], acc[i][j][r]);
      }
}

// ---------------------------------------------------------------------------
// Kernel 1b: softmax + max-subtract, stores attn'^T as bf16.
// attn'[c,d] = (1 - exp(S[c,d]-m_c)) / sum_d exp(S[c,d]-m_c)
// grid = BATCH, block = 128 (one score-row per thread).
// ---------------------------------------------------------------------------
__global__ __launch_bounds__(128) void softmax_kernel(
    const float* __restrict__ scores, unsigned short* __restrict__ attnT) {
  const int b   = blockIdx.x;
  const int row = threadIdx.x;                       // c index
  const float* s = scores + ((size_t)b * C_DIM + row) * C_DIM;
  float m = -3.4e38f;
  for (int c = 0; c < C_DIM; ++c) m = fmaxf(m, s[c]);
  float sum = 0.0f;
  for (int c = 0; c < C_DIM; ++c) sum += __expf(s[c] - m);
  const float inv = 1.0f / sum;
  unsigned short* at = attnT + (size_t)b * C_DIM * C_DIM;
  for (int c = 0; c < C_DIM; ++c) {
    const float a = (1.0f - __expf(s[c] - m)) * inv;
    at[c * C_DIM + row] = (unsigned short)f2bf(a);   // store transposed: [d][c]
  }
}

// ---------------------------------------------------------------------------
// Kernel 2: E = attn'^T . F  and  out = x + beta*E
// grid = (HW/NT, BATCH), block = 256 (8 waves). Wave w -> d-rows [16w,16w+16),
// 4 N-tiles of 16 within the 64-wide window.
// ---------------------------------------------------------------------------
__global__ __launch_bounds__(256) void apply_attn_kernel(
    const float* __restrict__ x, const float* __restrict__ beta_p,
    const unsigned short* __restrict__ attnT, float* __restrict__ out) {
  __shared__ unsigned short lds_at[C_DIM * LDSA];    // attn'^T : [d][c]
  __shared__ unsigned short lds_f[NT * LDSA];        // x chunk transposed: [n][c]
  const int b    = blockIdx.y;
  const int n0   = blockIdx.x * NT;
  const int tid  = threadIdx.x;
  const int lane = tid & 31;
  const int wave = tid >> 5;
  const int l15  = lane & 15;
  const int lhi  = lane >> 4;
  const float* xb = x + (size_t)b * C_DIM * HW;
  float*       ob = out + (size_t)b * C_DIM * HW;
  const unsigned* at32 = reinterpret_cast<const unsigned*>(attnT + (size_t)b * C_DIM * C_DIM);

  // load attn'^T (16384 bf16 = 8192 dwords), dword-wise
  #pragma unroll
  for (int i = 0; i < 32; ++i) {
    const int d   = i * 256 + tid;                   // dword index
    const int row = d >> 6;                          // 64 dwords per row
    const int col = (d & 63) * 2;
    *reinterpret_cast<unsigned*>(&lds_at[row * LDSA + col]) = at32[d];
  }
  // load 128 x 64 x-chunk, transpose into [n][c] as bf16
  {
    const int jj = (tid & 31) * 2;
    const int r8 = tid >> 5;
    #pragma unroll
    for (int pass = 0; pass < 16; ++pass) {
      const int c = pass * 8 + r8;
      const float2 d = *reinterpret_cast<const float2*>(xb + (size_t)c * HW + n0 + jj);
      lds_f[jj * LDSA + c]       = (unsigned short)f2bf(d.x);
      lds_f[(jj + 1) * LDSA + c] = (unsigned short)f2bf(d.y);
    }
  }
  __syncthreads();

  v8f acc[4];
  #pragma unroll
  for (int t = 0; t < 4; ++t)
    #pragma unroll
    for (int e = 0; e < 8; ++e) acc[t][e] = 0.0f;

  #pragma unroll
  for (int kk0 = 0; kk0 < C_DIM; kk0 += 32) {
    Frag A;
    #pragma unroll
    for (int v = 0; v < 8; ++v) {
      const int kk = kk0 + lhi * 8 + (v & 3) * 2 + (v >> 2) * 16;
      A.u[v] = *reinterpret_cast<const unsigned*>(&lds_at[(16 * wave + l15) * LDSA + kk]);
    }
    #pragma unroll
    for (int tn = 0; tn < 4; ++tn) {
      Frag Bf;
      #pragma unroll
      for (int v = 0; v < 8; ++v) {
        const int kk = kk0 + lhi * 8 + (v & 3) * 2 + (v >> 2) * 16;
        Bf.u[v] = *reinterpret_cast<const unsigned*>(&lds_f[(16 * tn + l15) * LDSA + kk]);
      }
      acc[tn] = __builtin_amdgcn_wmma_f32_16x16x32_bf16(
          false, A.v, false, Bf.v, (short)0, acc[tn], false, false);
    }
  }

  const float beta = beta_p[0];
  #pragma unroll
  for (int tn = 0; tn < 4; ++tn)
    #pragma unroll
    for (int r = 0; r < 8; ++r) {
      const int dch = 16 * wave + r + 8 * lhi;
      const int n   = n0 + 16 * tn + l15;
      const size_t idx = (size_t)dch * HW + n;
      ob[idx] = xb[idx] + beta * acc[tn][r];
    }
}

// ---------------------------------------------------------------------------
extern "C" void kernel_launch(void* const* d_in, const int* in_sizes, int n_in,
                              void* d_out, int out_size, void* d_ws, size_t ws_size,
                              hipStream_t stream) {
  const float* x    = (const float*)d_in[0];
  const float* beta = (const float*)d_in[1];
  float* out        = (float*)d_out;

  const size_t scores_bytes = (size_t)BATCH * C_DIM * C_DIM * sizeof(float); // 2 MiB
  float* scores         = (float*)d_ws;
  unsigned short* attnT = (unsigned short*)((char*)d_ws + scores_bytes);     // +1 MiB

  hipMemsetAsync(scores, 0, scores_bytes, stream);
  gram_partial_kernel<<<dim3(SEG, BATCH), 256, 0, stream>>>(x, scores);
  softmax_kernel<<<dim3(BATCH), 128, 0, stream>>>(scores, attnT);
  apply_attn_kernel<<<dim3(HW / NT, BATCH), 256, 0, stream>>>(x, beta, attnT, out);
}
